// FocusedLinearAttention_14551349199461
// MI455X (gfx1250) — compile-verified, matrix operand order checked by reading, not run on hardware
//
#include <hip/hip_runtime.h>
#include <hip/hip_bf16.h>

typedef __attribute__((ext_vector_type(16))) _Float16 v16h;
typedef __attribute__((ext_vector_type(8)))  _Float16 v8h;
typedef __attribute__((ext_vector_type(8)))  float    v8f;

static constexpr int DIMC  = 256;
static constexpr int NPIX  = 4096;   // 64*64
static constexpr int BATCH = 16;
static constexpr int BN    = BATCH * NPIX;   // 65536
static constexpr int NBH   = BATCH * 8;      // 128 batch-heads
static constexpr int HC    = 32;             // head channels

// ---- WMMA helpers -----------------------------------------------------------
static __device__ inline v8f wmma16(v16h a, v16h b, v8f c) {
  return __builtin_amdgcn_wmma_f32_16x16x32_f16(false, a, false, b, (short)0, c,
                                                false, false);
}

// A/B fragment for 16x16x32 f16 WMMA. Lane L holds row (L&15); K-halves map to
// two contiguous 8-half runs at K = 8*(L>>4) and K = 16 + 8*(L>>4).
static __device__ inline v16h load_frag(const _Float16* row, int half) {
  v8h lo = *reinterpret_cast<const v8h*>(row + half * 8);
  v8h hi = *reinterpret_cast<const v8h*>(row + 16 + half * 8);
  v16h r;
#pragma unroll
  for (int i = 0; i < 8; ++i) { r[i] = lo[i]; r[i + 8] = hi[i]; }
  return r;
}

static __device__ inline v8f zero8() {
  v8f z = {0.f, 0.f, 0.f, 0.f, 0.f, 0.f, 0.f, 0.f};
  return z;
}

// ---- K0: convert weights to f16 --------------------------------------------
__global__ void convert_weights(const float* __restrict__ qkv_w,
                                const float* __restrict__ proj_w,
                                _Float16* __restrict__ wq,
                                _Float16* __restrict__ wp) {
  int i = blockIdx.x * 256 + threadIdx.x;
  const int nq = 768 * 256, np = 256 * 256;
  for (; i < nq + np; i += gridDim.x * 256) {
    if (i < nq) wq[i] = (_Float16)qkv_w[i];
    else        wp[i - nq] = (_Float16)proj_w[i - nq];
  }
}

// ---- K1: softplus of scale param -------------------------------------------
__global__ void softplus_scale(const float* __restrict__ scale_p,
                               float* __restrict__ scale_s) {
  int c = threadIdx.x;
  float p = scale_p[c];
  scale_s[c] = (p > 20.f) ? p : log1pf(expf(p));
}

// ---- K2: x (B,C,N) f32 -> xh (B*N, C) f16, tiled transpose -----------------
__global__ void transpose_x(const float* __restrict__ x,
                            _Float16* __restrict__ xh) {
  __shared__ float tile[32][33];
  int n0 = blockIdx.x * 32, c0 = blockIdx.y * 32, bb = blockIdx.z;
  int tx = threadIdx.x & 31, ty = threadIdx.x >> 5;
#pragma unroll
  for (int i = 0; i < 4; ++i) {
    int c = c0 + ty + i * 8;
    tile[ty + i * 8][tx] = x[((size_t)bb * DIMC + c) * NPIX + n0 + tx];
  }
  __syncthreads();
#pragma unroll
  for (int i = 0; i < 4; ++i) {
    int n = n0 + ty + i * 8;
    xh[((size_t)bb * NPIX + n) * DIMC + c0 + tx] = (_Float16)tile[tx][ty + i * 8];
  }
}

// ---- K3: fused QKV GEMM (WMMA) + focusing nonlinearity ---------------------
// Block = 16 rows of (b,n); 8 waves x 6 column-tiles = 768 outputs, K=256.
__global__ void qkv_fused(const _Float16* __restrict__ xh,
                          const _Float16* __restrict__ wq,
                          const float* __restrict__ qkv_b,
                          const float* __restrict__ scale_s,
                          _Float16* __restrict__ q_h,   // (bh, n, c)
                          _Float16* __restrict__ k_t,   // (bh, c, n)
                          _Float16* __restrict__ v_t) { // (bh, c, n)
  __shared__ float raw[16][768];
  int tid = threadIdx.x, wave = tid >> 5, lane = tid & 31;
  int half = lane >> 4, l15 = lane & 15;
  int bn0 = blockIdx.x * 16;

  v8f acc[6];
#pragma unroll
  for (int t = 0; t < 6; ++t) {
    float bias = qkv_b[wave * 96 + t * 16 + l15];
#pragma unroll
    for (int r = 0; r < 8; ++r) acc[t][r] = bias;
  }
  const _Float16* arow = xh + (size_t)(bn0 + l15) * DIMC;
#pragma unroll
  for (int ks = 0; ks < 8; ++ks) {
    v16h a = load_frag(arow + ks * 32, half);
#pragma unroll
    for (int t = 0; t < 6; ++t) {
      const _Float16* brow = wq + (size_t)(wave * 96 + t * 16 + l15) * DIMC + ks * 32;
      v16h b = load_frag(brow, half);
      acc[t] = wmma16(a, b, acc[t]);
    }
  }
#pragma unroll
  for (int t = 0; t < 6; ++t)
#pragma unroll
    for (int r = 0; r < 8; ++r)
      raw[r + 8 * half][wave * 96 + t * 16 + l15] = acc[t][r];
  __syncthreads();

  // Nonlinearity: each half-wave (16 lanes) owns one row; lane ci covers
  // channels ci + 16*j (full 256-channel norms via 16-lane xor reduce).
  int row = 2 * wave + half;
  int bn = bn0 + row, bb = bn >> 12, n = bn & (NPIX - 1);
  float uq[16], uk[16], vv[16];
  float s2q = 0, s6q = 0, s2k = 0, s6k = 0;
#pragma unroll
  for (int j = 0; j < 16; ++j) {
    int c = l15 + j * 16;
    float sc = scale_s[c];
    float q = raw[row][c];       q = (fmaxf(q, 0.f) + 1e-6f) / sc;
    float k = raw[row][256 + c]; k = (fmaxf(k, 0.f) + 1e-6f) / sc;
    vv[j] = raw[row][512 + c];
    uq[j] = q; uk[j] = k;
    float q2 = q * q, k2 = k * k;
    s2q += q2; s6q += q2 * q2 * q2;
    s2k += k2; s6k += k2 * k2 * k2;
  }
  s2q += __shfl_xor(s2q, 1, 32); s2q += __shfl_xor(s2q, 2, 32);
  s2q += __shfl_xor(s2q, 4, 32); s2q += __shfl_xor(s2q, 8, 32);
  s6q += __shfl_xor(s6q, 1, 32); s6q += __shfl_xor(s6q, 2, 32);
  s6q += __shfl_xor(s6q, 4, 32); s6q += __shfl_xor(s6q, 8, 32);
  s2k += __shfl_xor(s2k, 1, 32); s2k += __shfl_xor(s2k, 2, 32);
  s2k += __shfl_xor(s2k, 4, 32); s2k += __shfl_xor(s2k, 8, 32);
  s6k += __shfl_xor(s6k, 1, 32); s6k += __shfl_xor(s6k, 2, 32);
  s6k += __shfl_xor(s6k, 4, 32); s6k += __shfl_xor(s6k, 8, 32);
  float fq = sqrtf(s2q / fmaxf(s6q, 1e-30f));
  float fk = sqrtf(s2k / fmaxf(s6k, 1e-30f));
#pragma unroll
  for (int j = 0; j < 16; ++j) {
    int c = l15 + j * 16;
    int bh = bb * 8 + (c >> 5), cc = c & 31;
    q_h[((size_t)bh * NPIX + n) * HC + cc]  = (_Float16)(uq[j] * uq[j] * uq[j] * fq);
    k_t[((size_t)bh * HC + cc) * NPIX + n]  = (_Float16)(uk[j] * uk[j] * uk[j] * fk);
    v_t[((size_t)bh * HC + cc) * NPIX + n]  = (_Float16)vv[j];
  }
}

// ---- K4: per-head kv = k^T v (32x32, K=4096) and ksum ----------------------
__global__ void kv_ksum(const _Float16* __restrict__ k_t,
                        const _Float16* __restrict__ v_t,
                        _Float16* __restrict__ kvT,    // (bh, d, c)
                        float* __restrict__ ksum) {    // (bh, c)
  __shared__ float kvacc[32][32];
  __shared__ float kss[32];
  int tid = threadIdx.x, wave = tid >> 5, lane = tid & 31;
  int half = lane >> 4, l15 = lane & 15;
  int bh = blockIdx.x;

  for (int i = tid; i < 1024; i += 256) kvacc[i >> 5][i & 31] = 0.f;
  if (tid < 32) kss[tid] = 0.f;
  __syncthreads();

  v8f acc[2][2] = {{zero8(), zero8()}, {zero8(), zero8()}};
  const _Float16* kb = k_t + (size_t)bh * HC * NPIX;
  const _Float16* vb = v_t + (size_t)bh * HC * NPIX;
#pragma unroll 4
  for (int ks = 0; ks < 16; ++ks) {
    int jb = wave * 512 + ks * 32;
    v16h a0 = load_frag(kb + (size_t)l15 * NPIX + jb, half);
    v16h a1 = load_frag(kb + (size_t)(16 + l15) * NPIX + jb, half);
    v16h b0 = load_frag(vb + (size_t)l15 * NPIX + jb, half);
    v16h b1 = load_frag(vb + (size_t)(16 + l15) * NPIX + jb, half);
    acc[0][0] = wmma16(a0, b0, acc[0][0]);
    acc[0][1] = wmma16(a0, b1, acc[0][1]);
    acc[1][0] = wmma16(a1, b0, acc[1][0]);
    acc[1][1] = wmma16(a1, b1, acc[1][1]);
  }
#pragma unroll
  for (int mt = 0; mt < 2; ++mt)
#pragma unroll
    for (int nt = 0; nt < 2; ++nt)
#pragma unroll
      for (int r = 0; r < 8; ++r)
        atomicAdd(&kvacc[mt * 16 + r + 8 * half][nt * 16 + l15], acc[mt][nt][r]);

  { // k column sums over N
    int c = tid & 31, seg = tid >> 5;
    const _Float16* p = kb + (size_t)c * NPIX + seg * 512;
    float s = 0.f;
    for (int j = 0; j < 512; ++j) s += (float)p[j];
    atomicAdd(&kss[c], s);
  }
  __syncthreads();
  for (int i = tid; i < 1024; i += 256) {
    int d = i >> 5, c = i & 31;
    kvT[(size_t)bh * 1024 + d * 32 + c] = (_Float16)kvacc[c][d];
  }
  if (tid < 32) ksum[bh * 32 + tid] = kss[tid];
}

// ---- K5: out = (q @ kv) * z, z = 1/(q . ksum + 1e-6) -----------------------
__global__ void q_kv_apply(const _Float16* __restrict__ q_h,
                           const _Float16* __restrict__ kvT,
                           const float* __restrict__ ksum,
                           _Float16* __restrict__ attn) { // (bn, C)
  __shared__ float zs[8][16];
  int tid = threadIdx.x, wave = tid >> 5, lane = tid & 31;
  int half = lane >> 4, l15 = lane & 15;
  int gw = blockIdx.x * 8 + wave;
  int bh = gw >> 8, rt = gw & 255, n0 = rt * 16;

  v16h a  = load_frag(q_h + ((size_t)bh * NPIX + n0 + l15) * HC, half);
  v16h b0 = load_frag(kvT + (size_t)bh * 1024 + (size_t)l15 * 32, half);
  v16h b1 = load_frag(kvT + (size_t)bh * 1024 + (size_t)(16 + l15) * 32, half);
  v8f acc0 = wmma16(a, b0, zero8());
  v8f acc1 = wmma16(a, b1, zero8());

  const float* ksp = ksum + bh * 32;
  float p = 0.f;
#pragma unroll
  for (int i = 0; i < 8; ++i) p += (float)a[i]     * ksp[half * 8 + i];
#pragma unroll
  for (int i = 0; i < 8; ++i) p += (float)a[i + 8] * ksp[16 + half * 8 + i];
  p += __shfl_xor(p, 16, 32);
  zs[wave][l15] = 1.f / (p + 1e-6f);
  __syncthreads();

  int bb = bh >> 3, hh = bh & 7;
#pragma unroll
  for (int r = 0; r < 8; ++r) {
    int n = n0 + r + 8 * half;
    float z = zs[wave][r + 8 * half];
    size_t rowb = ((size_t)bb * NPIX + n) * DIMC + hh * 32;
    attn[rowb + l15]      = (_Float16)(acc0[r] * z);
    attn[rowb + 16 + l15] = (_Float16)(acc1[r] * z);
  }
}

// ---- K6: depthwise 5x5 conv on v (per bh,c plane), accumulate into attn ----
__global__ void dwconv_add(const _Float16* __restrict__ v_t,
                           const float* __restrict__ dwc_w,
                           const float* __restrict__ dwc_b,
                           _Float16* __restrict__ attn) {
  int c = blockIdx.x & 31, bh = blockIdx.x >> 5;
  float w[25];
#pragma unroll
  for (int i = 0; i < 25; ++i) w[i] = dwc_w[c * 25 + i];
  float bias = dwc_b[c];
  const _Float16* vp = v_t + ((size_t)bh * HC + c) * NPIX;
  int bb = bh >> 3, hh = bh & 7;
#pragma unroll 2
  for (int it = 0; it < 16; ++it) {
    int pix = threadIdx.x + it * 256;
    int y = pix >> 6, x = pix & 63;
    float s = bias;
#pragma unroll
    for (int ky = 0; ky < 5; ++ky) {
      int yy = y + ky - 2;
      if (yy < 0 || yy > 63) continue;
#pragma unroll
      for (int kx = 0; kx < 5; ++kx) {
        int xx = x + kx - 2;
        if (xx < 0 || xx > 63) continue;
        s += w[ky * 5 + kx] * (float)vp[yy * 64 + xx];
      }
    }
    size_t idx = ((size_t)bb * NPIX + pix) * DIMC + hh * 32 + c;
    attn[idx] = (_Float16)((float)attn[idx] + s);
  }
}

// ---- K7: projection GEMM (BN x 256) @ (256 x 256)^T + bias, scatter NCHW ---
__global__ void proj_gemm(const _Float16* __restrict__ attn,
                          const _Float16* __restrict__ wp,
                          const float* __restrict__ proj_b,
                          float* __restrict__ out) { // (B, C, N)
  int tid = threadIdx.x, wave = tid >> 5, lane = tid & 31;
  int half = lane >> 4, l15 = lane & 15;
  int gw = blockIdx.x * 8 + wave;
  int mt = gw >> 4, nt = gw & 15;

  float bias = proj_b[nt * 16 + l15];
  v8f acc;
#pragma unroll
  for (int r = 0; r < 8; ++r) acc[r] = bias;
  const _Float16* arow = attn + (size_t)(mt * 16 + l15) * DIMC;
  const _Float16* brow = wp + (size_t)(nt * 16 + l15) * DIMC;
#pragma unroll
  for (int ks = 0; ks < 8; ++ks) {
    v16h a = load_frag(arow + ks * 32, half);
    v16h b = load_frag(brow + ks * 32, half);
    acc = wmma16(a, b, acc);
  }
#pragma unroll
  for (int r = 0; r < 8; ++r) {
    int bn = mt * 16 + r + 8 * half;
    int bb = bn >> 12, pix = bn & (NPIX - 1);
    int ch = nt * 16 + l15;
    out[((size_t)bb * DIMC + ch) * NPIX + pix] = acc[r];
  }
}

// ---- launch -----------------------------------------------------------------
extern "C" void kernel_launch(void* const* d_in, const int* in_sizes, int n_in,
                              void* d_out, int out_size, void* d_ws, size_t ws_size,
                              hipStream_t stream) {
  const float* x       = (const float*)d_in[0];
  const float* qkv_w   = (const float*)d_in[1];
  const float* qkv_b   = (const float*)d_in[2];
  const float* proj_w  = (const float*)d_in[3];
  const float* proj_b  = (const float*)d_in[4];
  const float* scale_p = (const float*)d_in[5];
  const float* dwc_w   = (const float*)d_in[6];
  const float* dwc_b   = (const float*)d_in[7];

  char* ws = (char*)d_ws;
  size_t o = 0;
  _Float16* xh    = (_Float16*)(ws + o); o += (size_t)BN * DIMC * 2;   // reused as attn
  _Float16* wq    = (_Float16*)(ws + o); o += 768 * 256 * 2;
  _Float16* wp    = (_Float16*)(ws + o); o += 256 * 256 * 2;
  float*    scs   = (float*)   (ws + o); o += 256 * 4 + 768;           // 1KB + 768 B pad
  _Float16* q_h   = (_Float16*)(ws + o); o += (size_t)BN * DIMC * 2;
  _Float16* k_t   = (_Float16*)(ws + o); o += (size_t)BN * DIMC * 2;
  _Float16* v_t   = (_Float16*)(ws + o); o += (size_t)BN * DIMC * 2;
  _Float16* kvT   = (_Float16*)(ws + o); o += (size_t)NBH * 1024 * 2;
  float*    ksm   = (float*)   (ws + o); o += (size_t)NBH * 32 * 4;
  _Float16* attn  = xh;  // xh dead after qkv_fused

  convert_weights<<<1024, 256, 0, stream>>>(qkv_w, proj_w, wq, wp);
  softplus_scale<<<1, 256, 0, stream>>>(scale_p, scs);
  transpose_x<<<dim3(128, 8, 16), 256, 0, stream>>>(x, xh);
  qkv_fused<<<BN / 16, 256, 0, stream>>>(xh, wq, qkv_b, scs, q_h, k_t, v_t);
  kv_ksum<<<NBH, 256, 0, stream>>>(k_t, v_t, kvT, ksm);
  q_kv_apply<<<(NBH * 256) / 8, 256, 0, stream>>>(q_h, kvT, ksm, attn);
  dwconv_add<<<NBH * 32, 256, 0, stream>>>(v_t, dwc_w, dwc_b, attn);
  proj_gemm<<<(BN / 16 * 16) / 8, 256, 0, stream>>>(attn, wp, proj_b, (float*)d_out);
}
